// MCMambaLM_31215822307447
// MI455X (gfx1250) — matrix operand order repaired and checked by reading, not compiled, not run on hardware
//
#include <hip/hip_runtime.h>

typedef __bf16 bf16;
typedef __attribute__((ext_vector_type(16))) __bf16 v16bf;
typedef __attribute__((ext_vector_type(8)))  float  v8f;
typedef __attribute__((ext_vector_type(4)))  unsigned int u32x4;
typedef __attribute__((ext_vector_type(8)))  int i32x8;
typedef __attribute__((ext_vector_type(4)))  int i32x4;

union FragBF { v16bf v; uint4 q[2]; };

#define T_LEN   2048
#define D_MODEL 768
#define DIN     1536
#define DSTATE  16
#define DTRANK  48
#define DTPAD   64
#define DBCW    80
#define NLAYER  4
#define SEGLEN  256
#define NSEG    8
#define KEMB    8
#define VVOC    1024

#if defined(__has_builtin)
#if __has_builtin(__builtin_amdgcn_tensor_load_to_lds) && \
    __has_builtin(__builtin_amdgcn_s_wait_tensorcnt)
#define USE_TDM 1
#endif
#endif
#ifndef USE_TDM
#define USE_TDM 0
#endif

// ---------------------------------------------------------------------------
// bf16 GEMM:  C[M,N] = A[M,K] * W[N,K]^T   (both operands K-contiguous)
// 128x128 block, K-step 32, 8 waves -> each wave 2x4 tiles of 16x16.
// Tiles staged into LDS by the Tensor Data Mover (double-buffered); hardware
// row padding gives an 80B LDS stride, hardware OOB clamp zero-fills N<128.
// ---------------------------------------------------------------------------
#define BM 128
#define BN 128
#define BK 32
#define LK 40   // padded LDS row stride (halfs): 64B data + 16B TDM pad

#if USE_TDM
__device__ __forceinline__ unsigned lds_off_u32(const bf16* p) {
  return (unsigned)(unsigned long long)(__attribute__((address_space(3))) bf16*)p;
}

// One 2D-tile descriptor: tile_dim0 = BK elements (64B rows), tile_dim1 = BM
// rows, row stride = K elements, LDS pad 4 DWORDs every 16 DWORDs.
__device__ __forceinline__ void tdm_load_tile(const bf16* gptr, unsigned lds_addr,
                                              int rows_rem, int K, int k_rem) {
  unsigned long long ga = (unsigned long long)gptr;
  u32x4 g0;
  g0[0] = 1u;                                             // count=1, user D#
  g0[1] = lds_addr;                                       // lds_addr (bytes)
  g0[2] = (unsigned)(ga & 0xFFFFFFFFu);                   // global_addr[31:0]
  g0[3] = (unsigned)((ga >> 32) & 0x1FFFFFFu) | (2u << 30); // addr[56:32]|type=2
  unsigned td0 = (unsigned)k_rem;                         // tensor_dim0 (OOB)
  unsigned td1 = (unsigned)rows_rem;                      // tensor_dim1 (OOB)
  i32x8 g1;
  g1[0] = (int)((1u << 16)      // data_size = 2 bytes
              | (1u << 20)      // pad_enable
              | (3u << 22)      // pad_interval: 16 DWORDs (64B)
              | (3u << 25));    // pad_amount:   4 DWORDs (16B)
  g1[1] = (int)((td0 & 0xFFFFu) << 16);                         // dim0 lo16
  g1[2] = (int)(((td0 >> 16) & 0xFFFFu) | ((td1 & 0xFFFFu) << 16));
  g1[3] = (int)(((td1 >> 16) & 0xFFFFu) | ((unsigned)BK << 16)); // tile_dim0
  g1[4] = (int)BM;                                              // tile_dim1
  g1[5] = (int)(unsigned)K;                                     // dim0_stride
  g1[6] = 0;
  g1[7] = 0;
  i32x4 gz = {0, 0, 0, 0};
#if defined(__clang_major__) && (__clang_major__ >= 23)
  i32x8 gz8 = {0, 0, 0, 0, 0, 0, 0, 0};
  __builtin_amdgcn_tensor_load_to_lds(g0, g1, gz, gz, gz8, 0);
#else
  __builtin_amdgcn_tensor_load_to_lds(g0, g1, gz, gz, 0);
#endif
}
#endif  // USE_TDM

__global__ __launch_bounds__(256)
void gemm_bf16_nt(const bf16* __restrict__ A, const bf16* __restrict__ W,
                  float* __restrict__ C, int M, int N, int K) {
  __shared__ bf16 As[2][BM * LK];
  __shared__ bf16 Bs[2][BN * LK];
  const int tid  = threadIdx.x;
  const int lane = tid & 31;
  const int wv   = tid >> 5;
  const int wm   = (wv >> 1) * 32;   // wave row base in block
  const int wn   = (wv & 1) * 64;    // wave col base in block
  const int m0   = blockIdx.y * BM;
  const int n0   = blockIdx.x * BN;

  v8f acc[2][4];
#pragma unroll
  for (int i = 0; i < 2; ++i)
#pragma unroll
    for (int j = 0; j < 4; ++j)
      acc[i][j] = (v8f){0.f,0.f,0.f,0.f,0.f,0.f,0.f,0.f};

  const int ksteps = K / BK;

#if USE_TDM
  const unsigned as_off0 = lds_off_u32(&As[0][0]);
  const unsigned as_off1 = lds_off_u32(&As[1][0]);
  const unsigned bs_off0 = lds_off_u32(&Bs[0][0]);
  const unsigned bs_off1 = lds_off_u32(&Bs[1][0]);
  if (wv == 0) {  // prologue: DMA tile 0 into buffer 0
    tdm_load_tile(A + (size_t)m0 * K, as_off0, M - m0, K, K);
    tdm_load_tile(W + (size_t)n0 * K, bs_off0, N - n0, K, K);
  }
#else
  const int lrow = tid >> 1;
  const int lcol = (tid & 1) << 4;
#endif

  for (int ks = 0; ks < ksteps; ++ks) {
    const int buf = ks & 1;
#if USE_TDM
    if (wv == 0) {
      if (ks + 1 < ksteps) {  // prefetch next tile into the other buffer
        const int k1 = (ks + 1) * BK;
        tdm_load_tile(A + (size_t)m0 * K + k1, buf ? as_off0 : as_off1,
                      M - m0, K, K - k1);
        tdm_load_tile(W + (size_t)n0 * K + k1, buf ? bs_off0 : bs_off1,
                      N - n0, K, K - k1);
        __builtin_amdgcn_s_wait_tensorcnt(2);  // current tile's 2 DMAs done
      } else {
        __builtin_amdgcn_s_wait_tensorcnt(0);
      }
    }
    __syncthreads();
#else
    {
      const int k0 = ks * BK;
      uint4 a0 = make_uint4(0u,0u,0u,0u), a1 = a0, b0 = a0, b1 = a0;
      if (m0 + lrow < M) {
        const uint4* p = (const uint4*)(A + (size_t)(m0 + lrow) * K + k0 + lcol);
        a0 = p[0]; a1 = p[1];
      }
      if (n0 + lrow < N) {
        const uint4* p = (const uint4*)(W + (size_t)(n0 + lrow) * K + k0 + lcol);
        b0 = p[0]; b1 = p[1];
      }
      *(uint4*)&As[buf][lrow * LK + lcol]     = a0;
      *(uint4*)&As[buf][lrow * LK + lcol + 8] = a1;
      *(uint4*)&Bs[buf][lrow * LK + lcol]     = b0;
      *(uint4*)&Bs[buf][lrow * LK + lcol + 8] = b1;
    }
    __syncthreads();
#endif

    const int fm  = lane & 15;
    const int hi  = lane >> 4;
    const int kbA = hi * 8;    // A frag: lanes<16 -> K 0..7 & 16..23; >=16 -> +8
    const int kbB = hi * 8;    // B frag: lanes<16 -> K 0..7 & 16..23; >=16 -> +8

    FragBF a[2], b[4];
#pragma unroll
    for (int mi = 0; mi < 2; ++mi) {
      const bf16* p = &As[buf][(wm + mi * 16 + fm) * LK + kbA];
      a[mi].q[0] = *(const uint4*)p;
      a[mi].q[1] = *(const uint4*)(p + 16);
    }
#pragma unroll
    for (int ni = 0; ni < 4; ++ni) {
      const bf16* p = &Bs[buf][(wn + ni * 16 + fm) * LK + kbB];
      b[ni].q[0] = *(const uint4*)p;
      b[ni].q[1] = *(const uint4*)(p + 16);
    }
#pragma unroll
    for (int mi = 0; mi < 2; ++mi)
#pragma unroll
      for (int ni = 0; ni < 4; ++ni)
        acc[mi][ni] = __builtin_amdgcn_wmma_f32_16x16x32_bf16(
            false, a[mi].v, false, b[ni].v, (short)0, acc[mi][ni], false, false);
    __syncthreads();
  }

  // D layout: VGPR r -> M = r (lanes 0-15) / 8+r (lanes 16-31); N = lane&15
  const int rowo = (lane >> 4) * 8;
  const int colL = lane & 15;
#pragma unroll
  for (int mi = 0; mi < 2; ++mi)
#pragma unroll
    for (int ni = 0; ni < 4; ++ni) {
      const int r0 = m0 + wm + mi * 16 + rowo;
      const int c  = n0 + wn + ni * 16 + colL;
      if (c < N) {
#pragma unroll
        for (int r = 0; r < 8; ++r)
          if (r0 + r < M) C[(size_t)(r0 + r) * N + c] = acc[mi][ni][r];
      }
    }
}

// ---------------------------------------------------------------------------
// Elementwise / small kernels
// ---------------------------------------------------------------------------
__global__ void f32_to_bf16_kernel(const float* __restrict__ in,
                                   bf16* __restrict__ out, int n) {
  int i = blockIdx.x * 256 + threadIdx.x;
  if (i < n) out[i] = (bf16)in[i];
}

__global__ void pad_dtw_kernel(const float* __restrict__ dt_w,
                               bf16* __restrict__ out) {
  int row = blockIdx.x;            // NLAYER*DIN
  int c   = threadIdx.x;           // 64
  out[(size_t)row * DTPAD + c] =
      (c < DTRANK) ? (bf16)dt_w[(size_t)row * DTRANK + c] : (bf16)0.f;
}

__global__ void pad_dtlow_kernel(const float* __restrict__ dbc,
                                 bf16* __restrict__ out) {
  int t = blockIdx.x;              // T
  int c = threadIdx.x;             // 64
  out[(size_t)t * DTPAD + c] =
      (c < DTRANK) ? (bf16)dbc[(size_t)t * DBCW + c] : (bf16)0.f;
}

__global__ void embed_kernel(const int* __restrict__ tokens,
                             const float* __restrict__ embed_w,
                             float* __restrict__ x) {
  int t = blockIdx.x;
  int d = blockIdx.y * 256 + threadIdx.x;
  float s = 0.f;
#pragma unroll
  for (int k = 0; k < KEMB; ++k) {
    int tok = tokens[t * KEMB + k];
    s += embed_w[((size_t)k * VVOC + tok) * D_MODEL + d];
  }
  x[(size_t)t * D_MODEL + d] = s;
}

__global__ void conv_silu_kernel(const float* __restrict__ xz,
                                 const float* __restrict__ conv_w,
                                 const float* __restrict__ conv_b,
                                 float* __restrict__ u, bf16* __restrict__ u_bf) {
  int c = blockIdx.x * 256 + threadIdx.x;   // DIN
  int t = blockIdx.y;                       // T
  float s = conv_b[c];
#pragma unroll
  for (int j = 0; j < 4; ++j) {
    int tt = t - 3 + j;
    if (tt >= 0) s += conv_w[c * 4 + j] * xz[(size_t)tt * (2 * DIN) + c];
  }
  float v = s / (1.f + __expf(-s));
  u[(size_t)t * DIN + c]    = v;
  u_bf[(size_t)t * DIN + c] = (bf16)v;
}

__global__ void delta_softplus_kernel(float* __restrict__ delta,
                                      const float* __restrict__ dt_b) {
  int c = blockIdx.x * 256 + threadIdx.x;   // DIN
  int t = blockIdx.y;
  float v = delta[(size_t)t * DIN + c] + dt_b[c];
  delta[(size_t)t * DIN + c] = (v > 20.f) ? v : __logf(1.f + __expf(v));
}

__global__ __launch_bounds__(128)
void scan_kernel(const float* __restrict__ delta, const float* __restrict__ u,
                 const float* __restrict__ dbc, const float* __restrict__ A_log,
                 float* __restrict__ ys) {
  __shared__ float sB[DSTATE], sC[DSTATE];
  int c = blockIdx.x * 128 + threadIdx.x;   // channel
  float A[DSTATE], h[DSTATE];
#pragma unroll
  for (int s = 0; s < DSTATE; ++s) {
    A[s] = -__expf(A_log[(size_t)c * DSTATE + s]);
    h[s] = 0.f;
  }
  for (int t = 0; t < T_LEN; ++t) {
    if (threadIdx.x < 32) {
      float v = dbc[(size_t)t * DBCW + DTRANK + threadIdx.x];
      if (threadIdx.x < DSTATE) sB[threadIdx.x] = v;
      else                      sC[threadIdx.x - DSTATE] = v;
    }
    __syncthreads();
    float d  = delta[(size_t)t * DIN + c];
    float du = d * u[(size_t)t * DIN + c];
    float y  = 0.f;
#pragma unroll
    for (int s = 0; s < DSTATE; ++s) {
      float dA = __expf(d * A[s]);
      h[s] = dA * h[s] + du * sB[s];
      y += h[s] * sC[s];
    }
    ys[(size_t)t * DIN + c] = y;
    __syncthreads();
  }
}

__global__ void ygate_kernel(const float* __restrict__ ys,
                             const float* __restrict__ u,
                             const float* __restrict__ xz,
                             const float* __restrict__ D_skip,
                             bf16* __restrict__ y_bf) {
  int c = blockIdx.x * 256 + threadIdx.x;
  int t = blockIdx.y;
  float z = xz[(size_t)t * (2 * DIN) + DIN + c];
  float y = (ys[(size_t)t * DIN + c] + D_skip[c] * u[(size_t)t * DIN + c]) *
            (z / (1.f + __expf(-z)));
  y_bf[(size_t)t * DIN + c] = (bf16)y;
}

__global__ void segstat_kernel(const float* __restrict__ x,
                               const float* __restrict__ x_out,
                               float* __restrict__ m_cur,
                               float* __restrict__ m_stk,
                               float* __restrict__ h_stk) {
  int d   = blockIdx.x * 256 + threadIdx.x;  // D_MODEL
  int seg = blockIdx.y;                      // NSEG
  int t0  = seg * SEGLEN;
  float run = 0.f;
  for (int i = 0; i < SEGLEN; ++i) {
    run += x[(size_t)(t0 + i) * D_MODEL + d];
    m_cur[(size_t)(t0 + i) * D_MODEL + d] = run / (float)(i + 1);
  }
  m_stk[(size_t)seg * D_MODEL + d] = run / (float)SEGLEN;
  h_stk[(size_t)seg * D_MODEL + d] = x_out[(size_t)(t0 + SEGLEN - 1) * D_MODEL + d];
}

__global__ __launch_bounds__(256)
void retattn_kernel(const float* __restrict__ u2, const float* __restrict__ m_stk,
                    const float* __restrict__ m_cur, const float* __restrict__ h_stk,
                    const float* __restrict__ x_out, float* __restrict__ x_new,
                    bf16* __restrict__ x_new_bf) {
  __shared__ float part[8][9];
  __shared__ float attnv[9];
  int t   = blockIdx.x;
  int tid = threadIdx.x;
  float p[9];
#pragma unroll
  for (int n = 0; n < 9; ++n) p[n] = 0.f;
  for (int d = tid; d < D_MODEL; d += 256) {
    float uv = u2[(size_t)t * D_MODEL + d];
#pragma unroll
    for (int n = 0; n < NSEG; ++n) p[n] += uv * m_stk[(size_t)n * D_MODEL + d];
    p[8] += uv * m_cur[(size_t)t * D_MODEL + d];
  }
#pragma unroll
  for (int n = 0; n < 9; ++n) {
    float v = p[n];
    for (int off = 16; off > 0; off >>= 1) v += __shfl_xor(v, off, 32);
    if ((tid & 31) == 0) part[tid >> 5][n] = v;
  }
  __syncthreads();
  if (tid == 0) {
    const float scale = 0.036084391824351615f;  // 1/sqrt(768)
    int seg = t >> 8;
    float sc[9], mx = -1e30f;
    for (int n = 0; n < 9; ++n) {
      float v = 0.f;
      for (int w = 0; w < 8; ++w) v += part[w][n];
      v *= scale;
      if (n < NSEG && !(seg > n)) v = -1e30f;
      sc[n] = v;
      if (v > mx) mx = v;
    }
    float den = 0.f;
    for (int n = 0; n < 9; ++n) { sc[n] = __expf(sc[n] - mx); den += sc[n]; }
    for (int n = 0; n < 9; ++n) attnv[n] = sc[n] / den;
  }
  __syncthreads();
  for (int d = tid; d < D_MODEL; d += 256) {
    float hr = 0.f;
#pragma unroll
    for (int n = 0; n < NSEG; ++n) hr += attnv[n] * h_stk[(size_t)n * D_MODEL + d];
    float v = attnv[8] * x_out[(size_t)t * D_MODEL + d] + hr;
    x_new[(size_t)t * D_MODEL + d]    = v;
    x_new_bf[(size_t)t * D_MODEL + d] = (bf16)v;
  }
}

// ---------------------------------------------------------------------------
extern "C" void kernel_launch(void* const* d_in, const int* in_sizes, int n_in,
                              void* d_out, int out_size, void* d_ws, size_t ws_size,
                              hipStream_t stream) {
  (void)in_sizes; (void)n_in; (void)out_size; (void)ws_size;
  const int*   tokens   = (const int*)  d_in[0];
  const float* embed_w  = (const float*)d_in[1];
  const float* W_u      = (const float*)d_in[2];
  const float* in_proj  = (const float*)d_in[3];
  const float* conv_w   = (const float*)d_in[4];
  const float* conv_b   = (const float*)d_in[5];
  const float* x_proj   = (const float*)d_in[6];
  const float* dt_w     = (const float*)d_in[7];
  const float* dt_b     = (const float*)d_in[8];
  const float* A_log    = (const float*)d_in[9];
  const float* D_skip   = (const float*)d_in[10];
  const float* out_proj = (const float*)d_in[11];
  const float* head_w   = (const float*)d_in[12];
  float* out = (float*)d_out;

  char* ws = (char*)d_ws;
  size_t off = 0;
  auto alloc = [&](size_t bytes) -> void* {
    void* p = ws + off;
    off += (bytes + 255) & ~(size_t)255;
    return p;
  };

  bf16* in_proj_bf  = (bf16*)alloc((size_t)NLAYER * 2 * DIN * D_MODEL * 2);
  bf16* out_proj_bf = (bf16*)alloc((size_t)NLAYER * D_MODEL * DIN * 2);
  bf16* W_u_bf      = (bf16*)alloc((size_t)NLAYER * D_MODEL * D_MODEL * 2);
  bf16* x_proj_bf   = (bf16*)alloc((size_t)NLAYER * DBCW * DIN * 2);
  bf16* dtw_pad_bf  = (bf16*)alloc((size_t)NLAYER * DIN * DTPAD * 2);
  bf16* head_w_bf   = (bf16*)alloc((size_t)KEMB * VVOC * D_MODEL * 2);

  float* x     = (float*)alloc((size_t)T_LEN * D_MODEL * 4);
  bf16*  x_bf  = (bf16*) alloc((size_t)T_LEN * D_MODEL * 2);
  float* xz    = (float*)alloc((size_t)T_LEN * 2 * DIN * 4);
  float* u     = (float*)alloc((size_t)T_LEN * DIN * 4);
  bf16*  u_bf  = (bf16*) alloc((size_t)T_LEN * DIN * 2);
  float* dbc   = (float*)alloc((size_t)T_LEN * DBCW * 4);
  bf16*  dtl_bf= (bf16*) alloc((size_t)T_LEN * DTPAD * 2);
  float* delta = (float*)alloc((size_t)T_LEN * DIN * 4);
  float* ys    = (float*)alloc((size_t)T_LEN * DIN * 4);
  bf16*  y_bf  = (bf16*) alloc((size_t)T_LEN * DIN * 2);
  float* x_out = (float*)alloc((size_t)T_LEN * D_MODEL * 4);
  float* u2    = (float*)alloc((size_t)T_LEN * D_MODEL * 4);
  float* m_cur = (float*)alloc((size_t)T_LEN * D_MODEL * 4);
  float* m_stk = (float*)alloc((size_t)NSEG * D_MODEL * 4);
  float* h_stk = (float*)alloc((size_t)NSEG * D_MODEL * 4);

  auto cvt = [&](const float* src, bf16* dst, size_t n) {
    f32_to_bf16_kernel<<<dim3((unsigned)((n + 255) / 256)), 256, 0, stream>>>(
        src, dst, (int)n);
  };
  cvt(in_proj,  in_proj_bf,  (size_t)NLAYER * 2 * DIN * D_MODEL);
  cvt(out_proj, out_proj_bf, (size_t)NLAYER * D_MODEL * DIN);
  cvt(W_u,      W_u_bf,      (size_t)NLAYER * D_MODEL * D_MODEL);
  cvt(x_proj,   x_proj_bf,   (size_t)NLAYER * DBCW * DIN);
  cvt(head_w,   head_w_bf,   (size_t)KEMB * VVOC * D_MODEL);
  pad_dtw_kernel<<<dim3(NLAYER * DIN), DTPAD, 0, stream>>>(dt_w, dtw_pad_bf);

  embed_kernel<<<dim3(T_LEN, D_MODEL / 256), 256, 0, stream>>>(tokens, embed_w, x);
  cvt(x, x_bf, (size_t)T_LEN * D_MODEL);

  auto gemm = [&](const bf16* A, const bf16* Wm, float* Cm, int M, int N, int K) {
    gemm_bf16_nt<<<dim3((N + BN - 1) / BN, (M + BM - 1) / BM), 256, 0, stream>>>(
        A, Wm, Cm, M, N, K);
  };

  for (int l = 0; l < NLAYER; ++l) {
    gemm(x_bf, in_proj_bf + (size_t)l * 2 * DIN * D_MODEL, xz, T_LEN, 2 * DIN, D_MODEL);
    conv_silu_kernel<<<dim3(DIN / 256, T_LEN), 256, 0, stream>>>(
        xz, conv_w + (size_t)l * DIN * 4, conv_b + (size_t)l * DIN, u, u_bf);
    gemm(u_bf, x_proj_bf + (size_t)l * DBCW * DIN, dbc, T_LEN, DBCW, DIN);
    pad_dtlow_kernel<<<dim3(T_LEN), DTPAD, 0, stream>>>(dbc, dtl_bf);
    gemm(dtl_bf, dtw_pad_bf + (size_t)l * DIN * DTPAD, delta, T_LEN, DIN, DTPAD);
    delta_softplus_kernel<<<dim3(DIN / 256, T_LEN), 256, 0, stream>>>(
        delta, dt_b + (size_t)l * DIN);
    scan_kernel<<<dim3(DIN / 128), 128, 0, stream>>>(
        delta, u, dbc, A_log + (size_t)l * DIN * DSTATE, ys);
    ygate_kernel<<<dim3(DIN / 256, T_LEN), 256, 0, stream>>>(
        ys, u, xz, D_skip + (size_t)l * DIN, y_bf);
    gemm(y_bf, out_proj_bf + (size_t)l * D_MODEL * DIN, x_out, T_LEN, D_MODEL, DIN);
    segstat_kernel<<<dim3(D_MODEL / 256, NSEG), 256, 0, stream>>>(
        x, x_out, m_cur, m_stk, h_stk);
    gemm(x_bf, W_u_bf + (size_t)l * D_MODEL * D_MODEL, u2, T_LEN, D_MODEL, D_MODEL);
    retattn_kernel<<<dim3(T_LEN), 256, 0, stream>>>(
        u2, m_stk, m_cur, h_stk, x_out, x, x_bf);
  }

  gemm(x_bf, head_w_bf, out, T_LEN, KEMB * VVOC, D_MODEL);
}
